// Diffeo_24567212933293
// MI455X (gfx1250) — matrix operand order repaired and checked by reading, not compiled, not run on hardware
//
#include <hip/hip_runtime.h>
#include <hip/hip_bf16.h>
#include <cmath>

// ---------------------------------------------------------------------------
// Diffeomorphic warp:
//   field = S * (c .* e) * S^T   (sine-mode synthesis, done with bf16 WMMA)
//   out   = bilinear_remap(img, x - dx, y - dy)
// n = 512, m = 100 (padded to K = 128), img (32,3,512,512).
// ---------------------------------------------------------------------------

typedef __attribute__((ext_vector_type(16))) __bf16 v16bf;
typedef __attribute__((ext_vector_type(8)))  __bf16 v8bf;
typedef __attribute__((ext_vector_type(8)))  float  v8f;

#define N_IMG   512
#define K_PAD   128
#define M_CUT   100
#define PI_F    3.14159265358979323846f

// ---- workspace layout (bytes) ----------------------------------------------
// S   : [512][128] bf16                       @ 0        (131072 B)
// At  : [2][128][128] bf16  (At[f][j][i])     @ 131072   ( 65536 B)
// M   : [2][512][128] bf16  (M[f][x][k])      @ 196608   (262144 B)
// tIdx: [512*512] u32   (yf<<16 | xf)         @ 458752   (1 MiB)
// tXv : [512*512] f32                         @ 1507328  (1 MiB)
// tYv : [512*512] f32                         @ 2555904  (1 MiB)
#define WS_S     0
#define WS_AT    131072
#define WS_M     196608
#define WS_TIDX  458752
#define WS_TXV   1507328
#define WS_TYV   2555904

// ---------------------------------------------------------------------------
// A-operand loader (16x32 bf16 tile):
//  lane L holds row M = L&15; vector element e holds
//  K = k0 + (L>=16 ? 16 : 0) + e
//  -> one contiguous 32 B load per lane.
__device__ __forceinline__ v16bf load_opA(const __bf16* __restrict__ base,
                                          int row, int k0, int lane) {
    const __bf16* p = base + row * K_PAD + k0 + ((lane >= 16) ? 16 : 0);
    return *(const v16bf*)(p);
}

// B-operand loader (32x16 bf16 tile): storage is K-contiguous per column
// (Bstore[col][k]).  Lane L holds col N = L&15; element e holds
// K = k0 + (L>=16 ? 16 : 0) + e  -> one contiguous 32 B load per lane.
__device__ __forceinline__ v16bf load_opB(const __bf16* __restrict__ base,
                                          int col, int k0, int lane) {
    const __bf16* p = base + col * K_PAD + k0 + ((lane >= 16) ? 16 : 0);
    return *(const v16bf*)(p);
}

// ---------------------------------------------------------------------------
// Setup: sine basis S[x][i] = sin(pi * x/(n-1) * (i+1)), zero-padded to 128.
__global__ void build_S_kernel(__bf16* __restrict__ S) {
    int t = blockIdx.x * blockDim.x + threadIdx.x;   // 0..65535
    int x = t >> 7;
    int i = t & 127;
    float v = 0.0f;
    if (i < M_CUT) {
        float xv = (float)x * (1.0f / (float)(N_IMG - 1));
        v = sinf(PI_F * xv * (float)(i + 1));
    }
    S[t] = (__bf16)v;
}

// Setup: At[f][j][i] = c_f[i][j] * e(i,j),  e = (r < m+0.5)/r, r=|(i+1,j+1)|.
__global__ void build_At_kernel(const float* __restrict__ cu,
                                const float* __restrict__ cv,
                                __bf16* __restrict__ At) {
    int t = blockIdx.x * blockDim.x + threadIdx.x;   // 0..16383
    int f = blockIdx.y;                              // 0=u, 1=v
    int j = t >> 7;
    int i = t & 127;
    float v = 0.0f;
    if (i < M_CUT && j < M_CUT) {
        float ki = (float)(i + 1), kj = (float)(j + 1);
        float r  = sqrtf(ki * ki + kj * kj);
        float e  = (r < (float)M_CUT + 0.5f) ? (1.0f / r) : 0.0f;
        const float* c = f ? cv : cu;
        v = c[i * M_CUT + j] * e;
    }
    At[f * (K_PAD * K_PAD) + t] = (__bf16)v;
}

// ---------------------------------------------------------------------------
// GEMM1:  M[f] = S (512x128) @ A[f] (128x128), stored as M[f][x][k] bf16.
// One wave = one 16x16 output tile.  grid = (32, 8, 2), block = 32.
__global__ void gemm1_kernel(const __bf16* __restrict__ S,
                             const __bf16* __restrict__ At,
                             __bf16* __restrict__ Mout) {
    int lane = threadIdx.x;
    int tm = blockIdx.x;          // row tile of S (x)
    int tn = blockIdx.y;          // col tile (j)
    int f  = blockIdx.z;          // field
    const __bf16* at = At   + f * (K_PAD * K_PAD);
    __bf16*       mo = Mout + f * (N_IMG * K_PAD);

    int arow = tm * 16 + (lane & 15);
    int bcol = tn * 16 + (lane & 15);

    v8f acc = {};
    #pragma unroll
    for (int k0 = 0; k0 < K_PAD; k0 += 32) {
        v16bf a = load_opA(S,  arow, k0, lane);
        v16bf b = load_opB(at, bcol, k0, lane);
        acc = __builtin_amdgcn_wmma_f32_16x16x32_bf16(
            false, a, false, b, (short)0, acc, false, false);
    }

    // C/D layout: element r -> (Mrow = r + 8*(lane>=16), Ncol = lane&15)
    int mbase = tm * 16 + ((lane >= 16) ? 8 : 0);
    int ncol  = tn * 16 + (lane & 15);
    #pragma unroll
    for (int r = 0; r < 8; ++r)
        mo[(mbase + r) * K_PAD + ncol] = (__bf16)acc[r];
}

// ---------------------------------------------------------------------------
// GEMM2 (fused): field[y][x] = sum_j S[y][j] * M[x][j] for BOTH fields, then
// directly emit per-pixel sample indices + bilinear weights.
// One wave = one 16x16 pixel tile.  grid = (32, 32), block = 32.
__global__ void gemm2_table_kernel(const __bf16* __restrict__ S,
                                   const __bf16* __restrict__ M,   // [2][512][128]
                                   unsigned* __restrict__ tIdx,
                                   float* __restrict__ tXv,
                                   float* __restrict__ tYv,
                                   float scale) {
    int lane = threadIdx.x;
    int tx = blockIdx.x;          // x tile
    int ty = blockIdx.y;          // y tile
    const __bf16* Mu = M;
    const __bf16* Mv = M + (N_IMG * K_PAD);

    int arow = ty * 16 + (lane & 15);   // y row of S
    int bcol = tx * 16 + (lane & 15);   // x col (rows of M storage)

    v8f au = {}, av = {};
    #pragma unroll
    for (int k0 = 0; k0 < K_PAD; k0 += 32) {
        v16bf a  = load_opA(S,  arow, k0, lane);
        v16bf bu = load_opB(Mu, bcol, k0, lane);
        v16bf bv = load_opB(Mv, bcol, k0, lane);
        au = __builtin_amdgcn_wmma_f32_16x16x32_bf16(
            false, a, false, bu, (short)0, au, false, false);
        av = __builtin_amdgcn_wmma_f32_16x16x32_bf16(
            false, a, false, bv, (short)0, av, false, false);
    }

    int ybase = ty * 16 + ((lane >= 16) ? 8 : 0);
    int x     = tx * 16 + (lane & 15);
    #pragma unroll
    for (int r = 0; r < 8; ++r) {
        int   y  = ybase + r;
        float dx = scale * au[r];
        float dy = scale * av[r];
        float xn = fminf(fmaxf((float)x - dx, 0.0f), (float)(N_IMG - 1));
        float yn = fminf(fmaxf((float)y - dy, 0.0f), (float)(N_IMG - 1));
        float xff = floorf(xn), yff = floorf(yn);
        int o = y * N_IMG + x;
        tIdx[o] = ((unsigned)(int)yff << 16) | (unsigned)(int)xff;
        tXv[o]  = xn - xff;
        tYv[o]  = yn - yff;
    }
}

// ---------------------------------------------------------------------------
// Remap: one thread per output pixel; table (3 MB) stays L2-resident and is
// reused across all 96 planes, so HBM traffic ~= img read + out write.
__global__ void remap_kernel(const float* __restrict__ img,
                             const unsigned* __restrict__ tIdx,
                             const float* __restrict__ tXv,
                             const float* __restrict__ tYv,
                             float* __restrict__ out) {
    int t     = blockIdx.x * blockDim.x + threadIdx.x;  // 0..25165823
    int pix   = t & (N_IMG * N_IMG - 1);                // y*512 + x
    int plane = t >> 18;                                // 0..95

    unsigned idx = tIdx[pix];
    float xv = tXv[pix];
    float yv = tYv[pix];
    int xf = (int)(idx & 0xFFFFu);
    int yf = (int)(idx >> 16);
    int xc = min(xf + 1, N_IMG - 1);
    int yc = min(yf + 1, N_IMG - 1);

    const float* p = img + (size_t)plane * (N_IMG * N_IMG);
    float v00 = p[yf * N_IMG + xf];
    float v01 = p[yf * N_IMG + xc];
    float v10 = p[yc * N_IMG + xf];
    float v11 = p[yc * N_IMG + xc];

    float top = v00 + xv * (v01 - v00);
    float bot = v10 + xv * (v11 - v10);
    out[(size_t)t] = top + yv * (bot - top);
}

// ---------------------------------------------------------------------------
extern "C" void kernel_launch(void* const* d_in, const int* in_sizes, int n_in,
                              void* d_out, int out_size, void* d_ws, size_t ws_size,
                              hipStream_t stream) {
    (void)in_sizes; (void)n_in; (void)out_size; (void)ws_size;

    const float* img = (const float*)d_in[0];
    const float* c_u = (const float*)d_in[1];
    const float* c_v = (const float*)d_in[2];
    float* out = (float*)d_out;

    char* ws = (char*)d_ws;
    __bf16*   S    = (__bf16*)  (ws + WS_S);
    __bf16*   At   = (__bf16*)  (ws + WS_AT);
    __bf16*   M    = (__bf16*)  (ws + WS_M);
    unsigned* tIdx = (unsigned*)(ws + WS_TIDX);
    float*    tXv  = (float*)   (ws + WS_TXV);
    float*    tYv  = (float*)   (ws + WS_TYV);

    // Deterministic scalar: T = midpoint of temperature_range(n=512, cut=100).
    const double n  = (double)N_IMG;
    const double m  = (double)M_CUT;
    const double lc = log(m + 1e-6);
    double T1 = 1.0 / (M_PI * n * n * lc);
    double T2 = 4.0 / (M_PI * M_PI * M_PI * m * m * lc);
    if (T2 < T1) T2 = T1;
    const double T = 0.5 * (T1 + T2);
    const float scale = (float)(sqrt(T) * n);   // dx = scale * field

    // 1) sine basis + scaled/transposed coefficient matrices (bf16, zero-padded)
    build_S_kernel<<<(N_IMG * K_PAD) / 256, 256, 0, stream>>>(S);
    build_At_kernel<<<dim3((K_PAD * K_PAD) / 256, 2), 256, 0, stream>>>(c_u, c_v, At);

    // 2) M[f] = S @ A[f]   (WMMA bf16, one wave per 16x16 tile)
    gemm1_kernel<<<dim3(N_IMG / 16, K_PAD / 16, 2), 32, 0, stream>>>(S, At, M);

    // 3) fields via S @ M[f]^T, fused into bilinear sample table
    gemm2_table_kernel<<<dim3(N_IMG / 16, N_IMG / 16), 32, 0, stream>>>(
        S, M, tIdx, tXv, tYv, scale);

    // 4) memory-bound remap of all 96 planes
    const int total = 32 * 3 * N_IMG * N_IMG;       // 25,165,824
    remap_kernel<<<total / 256, 256, 0, stream>>>(img, tIdx, tXv, tYv, out);
}
